// Performer_8426725835128
// MI455X (gfx1250) — matrix operand order repaired and checked by reading, not compiled, not run on hardware
//
#include <hip/hip_runtime.h>
#include <hip/hip_bf16.h>

// Performer FAVOR+ on MI455X (gfx1250, wave32, WMMA).
// GEMMs use V_WMMA_F32_16X16X32_F16 (f16 in, f32 accumulate) with
// register blocking: one wave owns a 16x64 (or 16x32) output tile so each
// A fragment feeds 4 (or 2) back-to-back WMMAs.
#define BATCH 8
#define SEQ   2048
#define DIM   256
#define MF    256   // number of random features (== DIM)

typedef __attribute__((ext_vector_type(16))) _Float16 v16h;
typedef __attribute__((ext_vector_type(8)))  _Float16 v8h;
typedef __attribute__((ext_vector_type(8)))  float    v8f;
typedef __attribute__((ext_vector_type(4)))  float    v4f;

// ---- WMMA fragment helpers (wave32; contiguous k order k = 16*(lane>>4) + s, same for A and B) ----
// A 16x32 f16: lane<16 holds row M=lane, K = [0..15];
//              lane>=16 holds row M=lane-16, K = [16..31].
// Caller passes p = rowPtr + k0 + (lane>>4)*16 ; the second chunk is p+8.
__device__ __forceinline__ v16h frag_a_f16(const _Float16* __restrict__ p) {
    v8h lo = *(const v8h*)(p);
    v8h hi = *(const v8h*)(p + 8);
    v16h r;
#pragma unroll
    for (int i = 0; i < 8; ++i) { r[i] = lo[i]; r[8 + i] = hi[i]; }
    return r;
}
__device__ __forceinline__ v16h frag_a_f32(const float* __restrict__ p) {
    v4f x0 = *(const v4f*)(p);      v4f x1 = *(const v4f*)(p + 4);
    v4f y0 = *(const v4f*)(p + 8); v4f y1 = *(const v4f*)(p + 12);
    v16h r;
#pragma unroll
    for (int i = 0; i < 4; ++i) {
        r[i]      = (_Float16)x0[i]; r[4 + i]  = (_Float16)x1[i];
        r[8 + i]  = (_Float16)y0[i]; r[12 + i] = (_Float16)y1[i];
    }
    return r;
}
// B 32x16 f16: lane holds column N=lane&15, K = (lane>>4)*16 + [0..15],
// contiguous when source is K-major. Caller passes p = colPtr + k0 + (lane>>4)*16.
__device__ __forceinline__ v16h frag_b_f16(const _Float16* __restrict__ p) {
    return *(const v16h*)(p);
}
__device__ __forceinline__ v16h frag_b_f32(const float* __restrict__ p) {
    v4f x0 = *(const v4f*)(p);     v4f x1 = *(const v4f*)(p + 4);
    v4f x2 = *(const v4f*)(p + 8); v4f x3 = *(const v4f*)(p + 12);
    v16h r;
#pragma unroll
    for (int i = 0; i < 4; ++i) {
        r[i]      = (_Float16)x0[i]; r[4 + i]  = (_Float16)x1[i];
        r[8 + i]  = (_Float16)x2[i]; r[12 + i] = (_Float16)x3[i];
    }
    return r;
}
__device__ __forceinline__ v8f wmma16x16x32(v16h a, v16h b, v8f c) {
    return __builtin_amdgcn_wmma_f32_16x16x32_f16(false, a, false, b, (short)0, c,
                                                  false, false);
}

// ---- 1) per-timestep Frobenius half-norms: hn[l] = 0.5*sqrt(sum_{b,d} X[b,l,d]^2)
__global__ void performer_norms(const float* __restrict__ Q,
                                const float* __restrict__ K,
                                float* __restrict__ hnQ, float* __restrict__ hnK) {
    const int l = blockIdx.x;
    const float* X = (blockIdx.y == 0) ? Q : K;
    float* hn      = (blockIdx.y == 0) ? hnQ : hnK;
    const int t = threadIdx.x;  // 0..255 == d
    float s = 0.f;
#pragma unroll
    for (int b = 0; b < BATCH; ++b) {
        float v = X[(size_t)b * SEQ * DIM + (size_t)l * DIM + t];
        s += v * v;
    }
    __shared__ float red[256];
    red[t] = s; __syncthreads();
    for (int off = 128; off > 0; off >>= 1) {
        if (t < off) red[t] += red[t + off];
        __syncthreads();
    }
    if (t == 0) hn[l] = 0.5f * sqrtf(red[0]);
}

// ---- 2) VT[b][d][l] = (f16) V[b][l][d]  (K-major layout for the S GEMM)
__global__ void performer_transpose_v(const float* __restrict__ V,
                                      _Float16* __restrict__ VT) {
    __shared__ float tile[32][33];
    const int b  = blockIdx.z;
    const int l0 = blockIdx.x * 32, d0 = blockIdx.y * 32;
    const int tx = threadIdx.x, ty = threadIdx.y;
    const float* Vb = V + (size_t)b * SEQ * DIM;
    _Float16* VTb   = VT + (size_t)b * DIM * SEQ;
#pragma unroll
    for (int k = 0; k < 32; k += 8)
        tile[ty + k][tx] = Vb[(size_t)(l0 + ty + k) * DIM + d0 + tx];
    __syncthreads();
#pragma unroll
    for (int k = 0; k < 32; k += 8)
        VTb[(size_t)(d0 + ty + k) * SEQ + l0 + tx] = (_Float16)tile[tx][ty + k];
}

// ---- 3) phi = exp(X @ proj^T - hn[l]) / sqrt(M).
// One wave = one 16x64 tile: A fragment reused across 4 column tiles.
// TR=false: write row-major phi[(b*L+l)*M + m] (for phiQ).
// TR=true : write transposed phiT[(b*M+m)*L + l] (for phiK; D-fragment's 8
//           consecutive rows become one contiguous 8xf16 store).
template <bool TR>
__global__ void performer_phi(const float* __restrict__ X,
                              const float* __restrict__ proj,
                              const float* __restrict__ hn,
                              _Float16* __restrict__ out) {
    const int lane = threadIdx.x & 31;
    const int wave = threadIdx.x >> 5;
    const int task = blockIdx.x * 8 + wave;       // (B*L/16) * (M/64) tasks
    const int cg = task & 3;                      // column group (64 cols)
    const int tr = task >> 2;                     // row tile over B*L
    const int r0 = tr * 16, c0 = cg * 64;
    const int half = lane >> 4, ln = lane & 15;

    const float* arow = X + (size_t)(r0 + ln) * DIM;          // row (b,l)
    const float* bcol[4];
#pragma unroll
    for (int j = 0; j < 4; ++j)
        bcol[j] = proj + (size_t)(c0 + j * 16 + ln) * DIM;    // proj[m][:] K-major

    v8f acc[4] = {};
#pragma unroll
    for (int k0 = 0; k0 < DIM; k0 += 32) {
        v16h a = frag_a_f32(arow + k0 + half * 16);
#pragma unroll
        for (int j = 0; j < 4; ++j)
            acc[j] = wmma16x16x32(a, frag_b_f32(bcol[j] + k0 + half * 16), acc[j]);
    }
    const float inv_sqrt_M = 0.0625f;             // 1/sqrt(256)
    const int rbase = r0 + half * 8;              // first of 8 rows in fragment
    const int l0 = rbase & (SEQ - 1);             // rows never straddle a batch
    if (TR) {
        const int b_ = rbase / SEQ;
#pragma unroll
        for (int j = 0; j < 4; ++j) {
            const int m = c0 + j * 16 + ln;
            v8h pack;
#pragma unroll
            for (int i = 0; i < 8; ++i)
                pack[i] = (_Float16)(__expf(acc[j][i] - hn[l0 + i]) * inv_sqrt_M);
            *(v8h*)(out + ((size_t)b_ * MF + m) * SEQ + l0) = pack;
        }
    } else {
#pragma unroll
        for (int j = 0; j < 4; ++j) {
            const int m = c0 + j * 16 + ln;
#pragma unroll
            for (int i = 0; i < 8; ++i)
                out[(size_t)(rbase + i) * MF + m] =
                    (_Float16)(__expf(acc[j][i] - hn[l0 + i]) * inv_sqrt_M);
        }
    }
}

// ---- 4) z[b][m] = sum_l phiKT[b][m][l]
__global__ void performer_z(const _Float16* __restrict__ phiKT,
                            float* __restrict__ z) {
    const int row = blockIdx.x;                   // b*M + m
    const int t = threadIdx.x;
    const _Float16* p = phiKT + (size_t)row * SEQ;
    float s = 0.f;
#pragma unroll
    for (int j = t; j < SEQ; j += 256) s += (float)p[j];
    __shared__ float red[256];
    red[t] = s; __syncthreads();
    for (int off = 128; off > 0; off >>= 1) {
        if (t < off) red[t] += red[t + off];
        __syncthreads();
    }
    if (t == 0) z[row] = red[0];
}

// ---- 5) S = phiK^T @ V per batch; stored transposed: ST[b][d][m] (f16).
// One wave = one 16x32 tile (2-wide blocking keeps 1024 waves in flight
// while halving A-fragment traffic). K = 2048 -> 64 k-steps.
__global__ void performer_s_gemm(const _Float16* __restrict__ phiKT,
                                 const _Float16* __restrict__ VT,
                                 _Float16* __restrict__ ST) {
    const int lane = threadIdx.x & 31;
    const int wave = threadIdx.x >> 5;
    const int task = blockIdx.x * 8 + wave;       // B * (M/16) * (D/32) tasks
    const int b  = task >> 7;                     // 128 tasks per batch
    const int t  = task & 127;
    const int tm = t >> 3, cg = t & 7;            // row tile over M, col group over D
    const int half = lane >> 4, ln = lane & 15;

    const _Float16* A = phiKT + (size_t)b * MF * SEQ + (size_t)(tm * 16 + ln) * SEQ;
    const _Float16* Bp[2];
#pragma unroll
    for (int j = 0; j < 2; ++j)
        Bp[j] = VT + (size_t)b * DIM * SEQ + (size_t)(cg * 32 + j * 16 + ln) * SEQ;

    v8f acc[2] = {};
#pragma unroll 8
    for (int k0 = 0; k0 < SEQ; k0 += 32) {
        v16h a = frag_a_f16(A + k0 + half * 16);
#pragma unroll
        for (int j = 0; j < 2; ++j)
            acc[j] = wmma16x16x32(a, frag_b_f16(Bp[j] + k0 + half * 16), acc[j]);
    }
#pragma unroll
    for (int j = 0; j < 2; ++j) {
        v8h pack;
#pragma unroll
        for (int i = 0; i < 8; ++i) pack[i] = (_Float16)acc[j][i];
        *(v8h*)(ST + ((size_t)b * DIM + cg * 32 + j * 16 + ln) * MF
                + tm * 16 + half * 8) = pack;
    }
}

// ---- 6) den[b,l] = phiQ[b,l,:] . z[b,:]
__global__ void performer_den(const _Float16* __restrict__ phiQ,
                              const float* __restrict__ z,
                              float* __restrict__ den) {
    const int r = blockIdx.x;                     // b*L + l
    const int b = r / SEQ;
    const int t = threadIdx.x;                    // == m
    float s = (float)phiQ[(size_t)r * MF + t] * z[b * MF + t];
    __shared__ float red[256];
    red[t] = s; __syncthreads();
    for (int off = 128; off > 0; off >>= 1) {
        if (t < off) red[t] += red[t + off];
        __syncthreads();
    }
    if (t == 0) den[r] = red[0];
}

// ---- 7) out = (phiQ @ S) / den. One wave = one 16x64 tile; the 1/den
// reciprocals are computed once per row-octet and reused across 4 col tiles.
__global__ void performer_out_gemm(const _Float16* __restrict__ phiQ,
                                   const _Float16* __restrict__ ST,
                                   const float* __restrict__ den,
                                   float* __restrict__ outp) {
    const int lane = threadIdx.x & 31;
    const int wave = threadIdx.x >> 5;
    const int task = blockIdx.x * 8 + wave;       // (B*L/16) * (D/64) tasks
    const int cg = task & 3;
    const int tr = task >> 2;                     // over B*L/16
    const int half = lane >> 4, ln = lane & 15;
    const int b = (tr * 16) / SEQ;

    const _Float16* A = phiQ + (size_t)(tr * 16 + ln) * MF;
    const _Float16* Bp[4];
#pragma unroll
    for (int j = 0; j < 4; ++j)
        Bp[j] = ST + (size_t)b * DIM * MF + (size_t)(cg * 64 + j * 16 + ln) * MF;

    v8f acc[4] = {};
#pragma unroll
    for (int k0 = 0; k0 < MF; k0 += 32) {
        v16h a = frag_a_f16(A + k0 + half * 16);
#pragma unroll
        for (int j = 0; j < 4; ++j)
            acc[j] = wmma16x16x32(a, frag_b_f16(Bp[j] + k0 + half * 16), acc[j]);
    }
    const int rbase = tr * 16 + half * 8;
    float id[8];
#pragma unroll
    for (int i = 0; i < 8; ++i) id[i] = 1.0f / den[rbase + i];
#pragma unroll
    for (int j = 0; j < 4; ++j) {
        const int d = cg * 64 + j * 16 + ln;
#pragma unroll
        for (int i = 0; i < 8; ++i)
            outp[(size_t)(rbase + i) * DIM + d] = acc[j][i] * id[i];
    }
}

extern "C" void kernel_launch(void* const* d_in, const int* in_sizes, int n_in,
                              void* d_out, int out_size, void* d_ws, size_t ws_size,
                              hipStream_t stream) {
    (void)in_sizes; (void)n_in; (void)out_size; (void)ws_size;
    const float* Q    = (const float*)d_in[0];
    const float* K    = (const float*)d_in[1];
    const float* V    = (const float*)d_in[2];
    /* d_in[3] = sent_embed_slice (unused in this path) */
    const float* proj = (const float*)d_in[4];
    float* outp = (float*)d_out;

    // carve scratch (256B-aligned segments)
    size_t off = 0;
    auto alloc = [&](size_t bytes) -> void* {
        void* p = (char*)d_ws + off;
        off += (bytes + 255) & ~(size_t)255;
        return p;
    };
    float*    hnQ   = (float*)alloc(SEQ * sizeof(float));
    float*    hnK   = (float*)alloc(SEQ * sizeof(float));
    float*    z     = (float*)alloc((size_t)BATCH * MF * sizeof(float));
    float*    den   = (float*)alloc((size_t)BATCH * SEQ * sizeof(float));
    _Float16* VT    = (_Float16*)alloc((size_t)BATCH * DIM * SEQ * 2);
    _Float16* phiQ  = (_Float16*)alloc((size_t)BATCH * SEQ * MF * 2);
    _Float16* phiKT = (_Float16*)alloc((size_t)BATCH * MF * SEQ * 2);
    _Float16* ST    = (_Float16*)alloc((size_t)BATCH * DIM * MF * 2);

    // 1) norms
    performer_norms<<<dim3(SEQ, 2), 256, 0, stream>>>(Q, K, hnQ, hnK);
    // 2) V transpose+convert
    performer_transpose_v<<<dim3(SEQ / 32, DIM / 32, BATCH), dim3(32, 8), 0, stream>>>(V, VT);
    // 3) phi GEMMs: (B*L/16)*(M/64) = 4096 wave-tasks, 8 waves/block
    performer_phi<false><<<4096 / 8, 256, 0, stream>>>(Q, proj, hnQ, phiQ);
    performer_phi<true ><<<4096 / 8, 256, 0, stream>>>(K, proj, hnK, phiKT);
    // 4) z
    performer_z<<<BATCH * MF, 256, 0, stream>>>(phiKT, z);
    // 5) S GEMM: B*(M/16)*(D/32) = 1024 wave-tasks
    performer_s_gemm<<<1024 / 8, 256, 0, stream>>>(phiKT, VT, ST);
    // 6) den
    performer_den<<<BATCH * SEQ, 256, 0, stream>>>(phiQ, z, den);
    // 7) out GEMM: (B*L/16)*(D/64) = 4096 wave-tasks
    performer_out_gemm<<<4096 / 8, 256, 0, stream>>>(phiQ, ST, den, outp);
}
